// ImgtoClass_Metric_75496935129606
// MI455X (gfx1250) — matrix operand order repaired and checked by reading, not compiled, not run on hardware
//
#include <hip/hip_runtime.h>

// ---- problem constants ----
#define WAY    5
#define SHOT   5
#define NK     3        // NEIGHBOR_K
#define NQ     75
#define CDIM   64
#define LQ     441      // H*W local descriptors
#define LPAD   448      // 28 * 16
#define QROWS  449      // 1 zero guard row (l = -1) + 448 padded rows
#define SSUP   2240     // SHOT * 448, shot-major padded support columns

typedef __attribute__((ext_vector_type(16))) _Float16 v16h;
typedef __attribute__((ext_vector_type(8)))  _Float16 v8h;
typedef __attribute__((ext_vector_type(8)))  float    v8f;

// ---------------------------------------------------------------------------
// WMMA fragment loaders (row-major [row][64] f16 operand storage)
// A 16x32 f16 layout: lane = 16*kh + m ; elems e -> K = 16*kh + e (one 32B chunk)
// B 32x16 f16 layout: lane = 16*kh + n ; elems e -> K = 16*kh + e (one 32B chunk)
// ---------------------------------------------------------------------------
__device__ __forceinline__ v16h load_fragA(const _Float16* rowp, int kstep, int kh) {
    return *(const v16h*)(rowp + kstep * 32 + kh * 16);
}
__device__ __forceinline__ v16h load_fragB(const _Float16* rowp, int kstep, int kh) {
    return *(const v16h*)(rowp + kstep * 32 + kh * 16);
}

// branch-free running top-3 insert (a >= b >= c), 5 v_min/v_max ops
__device__ __forceinline__ void ins3(float v, float& a, float& b, float& c) {
    const float x0 = fminf(a, v);
    a = fmaxf(a, v);
    const float x1 = fminf(b, x0);
    b = fmaxf(b, x0);
    c = fmaxf(c, x1);
}

// ---------------------------------------------------------------------------
// Kernel: zero output accumulators
// ---------------------------------------------------------------------------
__global__ void k_zero(float* __restrict__ p, int n) {
    int i = blockIdx.x * blockDim.x + threadIdx.x;
    if (i < n) p[i] = 0.0f;
}

// ---------------------------------------------------------------------------
// Kernel: L2-normalize query descriptors -> qn[q][row][c] f16
// row 0 is a zero guard (l = -1); rows 442..448 are zero pads.
// ---------------------------------------------------------------------------
__global__ void k_norm_q(const float* __restrict__ x1, _Float16* __restrict__ qn) {
    const int q   = blockIdx.x;
    const int row = blockIdx.y * 64 + threadIdx.x;
    if (row >= QROWS) return;
    _Float16* dst = qn + ((size_t)q * QROWS + row) * CDIM;
    const int l = row - 1;
    if (l >= 0 && l < LQ) {
        const float* src = x1 + (size_t)q * CDIM * LQ + l;
        float ss = 0.0f;
        for (int c = 0; c < CDIM; ++c) { float v = src[c * LQ]; ss += v * v; }
        const float rinv = rsqrtf(ss);
        for (int c = 0; c < CDIM; ++c) dst[c] = (_Float16)(src[c * LQ] * rinv);
    } else {
        for (int c = 0; c < CDIM; ++c) dst[c] = (_Float16)0.0f;
    }
}

// ---------------------------------------------------------------------------
// Kernel: L2-normalize support descriptors -> supn[way][shot*448 + l'][c] f16
// l' in [441,448) rows are zero pads (=> inner contribution 0 = conv SAME pad).
// ---------------------------------------------------------------------------
__global__ void k_norm_s(const float* __restrict__ x2, _Float16* __restrict__ supn) {
    const int img = blockIdx.x;                 // 0..24
    const int lp  = blockIdx.y * 64 + threadIdx.x;
    if (lp >= 448) return;
    const int way = img / SHOT, shot = img % SHOT;
    _Float16* dst = supn + ((size_t)way * SSUP + shot * 448 + lp) * CDIM;
    if (lp < LQ) {
        const float* src = x2 + (size_t)img * CDIM * LQ + lp;
        float ss = 0.0f;
        for (int c = 0; c < CDIM; ++c) { float v = src[c * LQ]; ss += v * v; }
        const float rinv = rsqrtf(ss);
        for (int c = 0; c < CDIM; ++c) dst[c] = (_Float16)(src[c * LQ] * rinv);
    } else {
        for (int c = 0; c < CDIM; ++c) dst[c] = (_Float16)0.0f;
    }
}

// ---------------------------------------------------------------------------
// Kernel: class prototypes  proto[w][c] = mean_s supn[w][s][c]  (pads are 0)
// ---------------------------------------------------------------------------
__global__ void k_proto(const _Float16* __restrict__ supn, float* __restrict__ proto) {
    const int w = blockIdx.x, c = threadIdx.x;
    float s = 0.0f;
    for (int n = 0; n < SSUP; ++n) s += (float)supn[((size_t)w * SSUP + n) * CDIM + c];
    proto[w * CDIM + c] = s / (float)(SHOT * LQ);
}

// ---------------------------------------------------------------------------
// Kernel: support descriptor weights
// sw[w][s] = sigmoid( (sum_p Wsup[p]*proto[p]) . sup[w][:,s] + bsup )
// ---------------------------------------------------------------------------
__global__ void k_sw(const _Float16* __restrict__ supn, const float* __restrict__ proto,
                     const float* __restrict__ Wsup, const float* __restrict__ bsup,
                     float* __restrict__ swb) {
    __shared__ float vsup[CDIM];
    const int w = blockIdx.y;
    if (threadIdx.x < CDIM) {
        float a = 0.0f;
        for (int p = 0; p < WAY; ++p) a += Wsup[p] * proto[p * CDIM + threadIdx.x];
        vsup[threadIdx.x] = a;
    }
    __syncthreads();
    const int n = blockIdx.x * 256 + threadIdx.x;
    if (n >= SSUP) return;
    const int lp = n % 448;
    float o = 0.0f;
    if (lp < LQ) {
        const _Float16* sp = supn + ((size_t)w * SSUP + n) * CDIM;
        float d = 0.0f;
        for (int c = 0; c < CDIM; ++c) d += vsup[c] * (float)sp[c];
        o = 1.0f / (1.0f + expf(-(d + bsup[0])));
    }
    swb[w * SSUP + n] = o;
}

// ---------------------------------------------------------------------------
// Kernel 1 (heavy): fused GEMM + shot-max + 3x3 conv + top-3 -> relation[q][w][l]
// Block = (l-tile of 64 rows + 1-row halo each side, all 448 support l' cols).
// 10 waves: wave = 2*rowtile + colhalf. Shot-max is 5 accumulator fmax's.
// All 10 B-fragment loads are hoisted ahead of the 10 WMMAs so the compiler
// can clause them; the map-row guard is a branch-free clamp to scratch row 67
// (conv only reads rows 0..65), avoiding exec-mask churn in the hot loop.
// ---------------------------------------------------------------------------
__global__ __launch_bounds__(320)
void k_sim_relation(const _Float16* __restrict__ qn, const _Float16* __restrict__ supn,
                    const float* __restrict__ convk, const float* __restrict__ convb,
                    float* __restrict__ relation) {
    __shared__ _Float16 map[68 * 448];
    __shared__ float    buf[64 * 5 * 3];

    const int lt = blockIdx.x;               // 0..6
    const int w  = blockIdx.y;               // 0..4
    const int q  = blockIdx.z;               // 0..74
    const int l0 = lt * 64;
    const int tid  = threadIdx.x;
    const int wid  = tid >> 5;
    const int lane = tid & 31;
    const int m16  = lane & 15;
    const int kh   = lane >> 4;
    const int rt   = wid >> 1;               // row tile 0..4 (80 rows, halo)
    const int ch   = wid & 1;                // column half

    // Preload A fragments (reused across all 14 column iterations)
    int arow = l0 - 1 + rt * 16 + m16;       // -1 .. 462
    int aidx = arow + 1; if (aidx > 448) aidx = 448;   // clamp to a zero row
    const _Float16* arp = qn + ((size_t)q * QROWS + aidx) * CDIM;
    const v16h a0 = load_fragA(arp, 0, kh);
    const v16h a1 = load_fragA(arp, 1, kh);

    const _Float16* supw = supn + (size_t)w * SSUP * CDIM;

    for (int it = 0; it < 14; ++it) {
        const int lp = it * 32 + ch * 16 + m16;     // support l' column 0..447

        // hoist all B fragments (5 shots x 2 ksteps) -> one long load clause
        v16h b0s[SHOT], b1s[SHOT];
#pragma unroll
        for (int shot = 0; shot < SHOT; ++shot) {
            const _Float16* brp = supw + ((size_t)shot * 448 + lp) * CDIM;
            b0s[shot] = load_fragB(brp, 0, kh);
            b1s[shot] = load_fragB(brp, 1, kh);
        }

        float mx[8];
#pragma unroll
        for (int j = 0; j < 8; ++j) mx[j] = -3.0e38f;
#pragma unroll
        for (int shot = 0; shot < SHOT; ++shot) {
            v8f acc = {};
            acc = __builtin_amdgcn_wmma_f32_16x16x32_f16(false, a0, false, b0s[shot],
                                                         (short)0, acc, false, false);
            acc = __builtin_amdgcn_wmma_f32_16x16x32_f16(false, a1, false, b1s[shot],
                                                         (short)0, acc, false, false);
#pragma unroll
            for (int j = 0; j < 8; ++j) mx[j] = fmaxf(mx[j], acc[j]);
        }
#pragma unroll
        for (int j = 0; j < 8; ++j) {
            const int relrow = rt * 16 + kh * 8 + j;        // C row = 8*half + j
            const int rstore = relrow < 68 ? relrow : 67;   // clamp to scratch row
            map[rstore * 448 + lp] = (_Float16)mx[j];
        }
    }
    __syncthreads();

    // ---- 3x3 conv (cross-correlation, SAME) + per-row top-3, from LDS ----
    // Sliding 3x3 window: only 3 new LDS loads per output column.
    float ck[9];
#pragma unroll
    for (int i = 0; i < 9; ++i) ck[i] = convk[i];
    const float cb = convb[0];

    const int r    = tid & 63;           // output row within tile
    const int part = tid >> 6;           // 0..4 column strip
    const int l    = l0 + r;
    float t0 = -3.0e38f, t1 = -3.0e38f, t2 = -3.0e38f;
    if (l < LQ) {
        const int rr = r + 1;            // map relrow of output row
        const _Float16* m0 = &map[(rr - 1) * 448];
        const _Float16* m1 = &map[rr * 448];
        const _Float16* m2 = &map[(rr + 1) * 448];
        const int c0 = part * 90;
        int c1 = c0 + 90; if (c1 > LQ) c1 = LQ;

        float wA0, wA1, wA2, wB0, wB1, wB2;
        if (c0 > 0) {
            wA0 = (float)m0[c0 - 1]; wA1 = (float)m1[c0 - 1]; wA2 = (float)m2[c0 - 1];
        } else {
            wA0 = 0.0f; wA1 = 0.0f; wA2 = 0.0f;     // SAME zero pad at l' = -1
        }
        wB0 = (float)m0[c0]; wB1 = (float)m1[c0]; wB2 = (float)m2[c0];

        for (int col = c0; col < c1; ++col) {
            // col+1 <= 441 and map column 441 is exactly zero (SAME pad)
            const float wC0 = (float)m0[col + 1];
            const float wC1 = (float)m1[col + 1];
            const float wC2 = (float)m2[col + 1];
            float a = cb;
            a += ck[0] * wA0 + ck[1] * wB0 + ck[2] * wC0;
            a += ck[3] * wA1 + ck[4] * wB1 + ck[5] * wC1;
            a += ck[6] * wA2 + ck[7] * wB2 + ck[8] * wC2;
            ins3(a, t0, t1, t2);
            wA0 = wB0; wA1 = wB1; wA2 = wB2;
            wB0 = wC0; wB1 = wC1; wB2 = wC2;
        }
    }
    float* bp = &buf[(r * 5 + part) * 3];
    bp[0] = t0; bp[1] = t1; bp[2] = t2;
    __syncthreads();

    if (tid < 64) {
        const int ll = l0 + tid;
        if (ll < LQ) {
            float a0c = -3.0e38f, a1c = -3.0e38f, a2c = -3.0e38f;
            for (int p = 0; p < 5; ++p) {
                const float* sp = &buf[(tid * 5 + p) * 3];
                ins3(sp[0], a0c, a1c, a2c);
                ins3(sp[1], a0c, a1c, a2c);
                ins3(sp[2], a0c, a1c, a2c);
            }
            relation[((size_t)q * WAY + w) * LPAD + ll] = a0c + a1c + a2c;
        }
    }
}

// ---------------------------------------------------------------------------
// Kernel: query descriptor weights wq[q][l] = sigmoid(relation . Ww + bw)
// ---------------------------------------------------------------------------
__global__ void k_wq(const float* __restrict__ rel, const float* __restrict__ Ww,
                     const float* __restrict__ bw, float* __restrict__ wq) {
    const int idx = blockIdx.x * 256 + threadIdx.x;
    if (idx >= NQ * LPAD) return;
    const int q = idx / LPAD, l = idx % LPAD;
    float o = 0.0f;
    if (l < LQ) {
        float a = 0.0f;
        for (int w = 0; w < WAY; ++w) a += Ww[w] * rel[((size_t)q * WAY + w) * LPAD + l];
        o = 1.0f / (1.0f + expf(-(a + bw[0])));
    }
    wq[idx] = o;
}

// ---------------------------------------------------------------------------
// Kernel 2 (heavy): fused GEMM + sw-scale + running top-3 + wq-weighted sum.
// out[q][w] += sum_l wq[q][l] * top3_s( sw[w][s] * inner[q][w][l][s] )
// (valid because wq > 0 commutes with top-k). Per-lane top-3 over its column
// subset, shuffle-merged across the 16 lanes holding each row.
// ---------------------------------------------------------------------------
__global__ __launch_bounds__(256)
void k_score(const _Float16* __restrict__ qn, const _Float16* __restrict__ supn,
             const float* __restrict__ swb, const float* __restrict__ wqb,
             float* __restrict__ out) {
    __shared__ float cand[64][2][3];
    __shared__ float partial[64];

    const int lt = blockIdx.x, w = blockIdx.y, q = blockIdx.z;
    const int l0 = lt * 64;
    const int tid  = threadIdx.x;
    const int wid  = tid >> 5;
    const int lane = tid & 31;
    const int m16  = lane & 15;
    const int kh   = lane >> 4;
    const int rt   = wid >> 1;     // 0..3 (64 rows)
    const int ch   = wid & 1;

    const int arow = l0 + rt * 16 + m16;                 // 0..447 (pads are zero)
    const _Float16* arp = qn + ((size_t)q * QROWS + arow + 1) * CDIM;
    const v16h a0 = load_fragA(arp, 0, kh);
    const v16h a1 = load_fragA(arp, 1, kh);

    const _Float16* supw = supn + (size_t)w * SSUP * CDIM;
    const float*    sww  = swb + w * SSUP;

    float top[8][3];
#pragma unroll
    for (int j = 0; j < 8; ++j) top[j][0] = top[j][1] = top[j][2] = -3.0e38f;

    for (int it = 0; it < SSUP / 32; ++it) {
        const int s = it * 32 + ch * 16 + m16;
        const _Float16* brp = supw + (size_t)s * CDIM;
        const v16h b0 = load_fragB(brp, 0, kh);
        const v16h b1 = load_fragB(brp, 1, kh);
        v8f acc = {};
        acc = __builtin_amdgcn_wmma_f32_16x16x32_f16(false, a0, false, b0,
                                                     (short)0, acc, false, false);
        acc = __builtin_amdgcn_wmma_f32_16x16x32_f16(false, a1, false, b1,
                                                     (short)0, acc, false, false);
        const float swv = sww[s];
        // exclude zero-pad columns (l' >= 441) branch-free
        const float gate = ((s % 448) < LQ) ? 0.0f : -3.0e38f;
#pragma unroll
        for (int j = 0; j < 8; ++j) {
            const float v = acc[j] * swv + gate;
            ins3(v, top[j][0], top[j][1], top[j][2]);
        }
    }

    // merge top-3 across the 16 lanes sharing each row (xor stays within half)
    for (int mask = 1; mask <= 8; mask <<= 1) {
#pragma unroll
        for (int j = 0; j < 8; ++j) {
            const float o0 = __shfl_xor(top[j][0], mask);
            const float o1 = __shfl_xor(top[j][1], mask);
            const float o2 = __shfl_xor(top[j][2], mask);
            ins3(o0, top[j][0], top[j][1], top[j][2]);
            ins3(o1, top[j][0], top[j][1], top[j][2]);
            ins3(o2, top[j][0], top[j][1], top[j][2]);
        }
    }
    if (m16 == 0) {
#pragma unroll
        for (int j = 0; j < 8; ++j) {
            const int r = rt * 16 + kh * 8 + j;
            cand[r][ch][0] = top[j][0];
            cand[r][ch][1] = top[j][1];
            cand[r][ch][2] = top[j][2];
        }
    }
    __syncthreads();

    if (tid < 64) {
        const int l = l0 + tid;
        float p = 0.0f;
        if (l < LQ) {
            float a0c = cand[tid][0][0], a1c = cand[tid][0][1], a2c = cand[tid][0][2];
            ins3(cand[tid][1][0], a0c, a1c, a2c);
            ins3(cand[tid][1][1], a0c, a1c, a2c);
            ins3(cand[tid][1][2], a0c, a1c, a2c);
            p = (a0c + a1c + a2c) * wqb[(size_t)q * LPAD + l];
        }
        partial[tid] = p;
    }
    __syncthreads();
    if (tid == 0) {
        float s = 0.0f;
        for (int i = 0; i < 64; ++i) s += partial[i];
        atomicAdd(&out[q * WAY + w], s);
    }
}

// ---------------------------------------------------------------------------
extern "C" void kernel_launch(void* const* d_in, const int* in_sizes, int n_in,
                              void* d_out, int out_size, void* d_ws, size_t ws_size,
                              hipStream_t stream) {
    (void)in_sizes; (void)n_in; (void)out_size; (void)ws_size;
    const float* x1    = (const float*)d_in[0];
    const float* x2    = (const float*)d_in[1];
    const float* Wsup  = (const float*)d_in[2];
    const float* bsup  = (const float*)d_in[3];
    const float* convk = (const float*)d_in[4];
    const float* convb = (const float*)d_in[5];
    const float* Ww    = (const float*)d_in[6];
    const float* bw    = (const float*)d_in[7];
    float* out = (float*)d_out;

    char* ws = (char*)d_ws;
    auto alloc = [&](size_t bytes) -> char* {
        char* p = ws;
        ws += (bytes + 255) & ~(size_t)255;
        return p;
    };
    _Float16* qn    = (_Float16*)alloc((size_t)NQ * QROWS * CDIM * 2);   // ~4.3 MB
    _Float16* supn  = (_Float16*)alloc((size_t)WAY * SSUP * CDIM * 2);   // ~1.4 MB
    float*    proto = (float*)alloc((size_t)WAY * CDIM * 4);
    float*    swb   = (float*)alloc((size_t)WAY * SSUP * 4);
    float*    rel   = (float*)alloc((size_t)NQ * WAY * LPAD * 4);        // ~0.7 MB
    float*    wqb   = (float*)alloc((size_t)NQ * LPAD * 4);

    k_zero<<<1, 512, 0, stream>>>(out, NQ * WAY);
    k_norm_q<<<dim3(NQ, 8), 64, 0, stream>>>(x1, qn);
    k_norm_s<<<dim3(WAY * SHOT, 7), 64, 0, stream>>>(x2, supn);
    k_proto<<<WAY, CDIM, 0, stream>>>(supn, proto);
    k_sw<<<dim3((SSUP + 255) / 256, WAY), 256, 0, stream>>>(supn, proto, Wsup, bsup, swb);
    k_sim_relation<<<dim3(7, WAY, NQ), 320, 0, stream>>>(qn, supn, convk, convb, rel);
    k_wq<<<(NQ * LPAD + 255) / 256, 256, 0, stream>>>(rel, Ww, bw, wqb);
    k_score<<<dim3(7, WAY, NQ), 256, 0, stream>>>(qn, supn, swb, wqb, out);
}
